// MixtralMoE_62397284876806
// MI455X (gfx1250) — matrix operand order repaired and checked by reading, not compiled, not run on hardware
//
#include <hip/hip_runtime.h>
#include <hip/hip_bf16.h>

typedef float  v8f   __attribute__((ext_vector_type(8)));
typedef __bf16 v8bf  __attribute__((ext_vector_type(8)));
typedef __bf16 v16bf __attribute__((ext_vector_type(16)));

#define NEXP 8      // experts
#define TC   1024   // token-slot chunk per expert
#define BM   128    // block tile M (token slots)
#define BN   64     // block tile N
#define BK   32     // WMMA K step (bf16)
#define LDP  40     // LDS row stride in halves (32 data + 8 pad -> conflict-free)

// ---------------------------------------------------------------------------
// Optional CDNA5 async global->LDS copy (bypasses VGPRs, tracked by ASYNCcnt)
// ---------------------------------------------------------------------------
#if defined(__HIP_DEVICE_COMPILE__) && __has_builtin(__builtin_amdgcn_global_load_async_to_lds_b128)
#define USE_ASYNC_LDS 1
#endif

#ifdef USE_ASYNC_LDS
typedef int v4i_gcc __attribute__((vector_size(16)));
typedef __attribute__((address_space(1))) v4i_gcc g_v4i;
typedef __attribute__((address_space(3))) v4i_gcc l_v4i;
__device__ __forceinline__ void async_ld16(void* lds, const void* g) {
  __builtin_amdgcn_global_load_async_to_lds_b128(
      (g_v4i*)(uintptr_t)g, (l_v4i*)(unsigned)(uintptr_t)lds, 0, 0);
}
__device__ __forceinline__ void async_fence() {
#if __has_builtin(__builtin_amdgcn_s_wait_asynccnt)
  __builtin_amdgcn_s_wait_asynccnt(0);
#else
  asm volatile("s_wait_asynccnt 0" ::: "memory");
#endif
}
#endif

// ---------------------------------------------------------------------------
// Fragment loaders from LDS, matching CDNA5 16-bit WMMA VGPR layouts.
// A (16x32, M x K): lane<16 -> row=lane, K = [0..7] + [16..23]
//                   lane>=16 -> row=lane-16, K = [8..15] + [24..31]
// B (32x16, K x N): lane<16 -> col=lane, K = [0..7] + [16..23]
//                   lane>=16 -> col=lane-16, K = [8..15] + [24..31]
// ---------------------------------------------------------------------------
__device__ __forceinline__ v16bf load_a_frag(const __bf16* tile, int lane) {
  int r  = lane & 15;
  int kA = (lane < 16) ? 0 : 8;
  const __bf16* p = tile + r * LDP + kA;
  v8bf lo = *(const v8bf*)p;
  v8bf hi = *(const v8bf*)(p + 16);
  return __builtin_shufflevector(lo, hi, 0,1,2,3,4,5,6,7,8,9,10,11,12,13,14,15);
}

__device__ __forceinline__ v16bf load_b_frag(const __bf16* tile, int lane) {
  int c  = lane & 15;
  int kB = (lane < 16) ? 0 : 8;
  const __bf16* p = tile + c * LDP + kB;
  v8bf lo = *(const v8bf*)p;
  v8bf hi = *(const v8bf*)(p + 16);
  return __builtin_shufflevector(lo, hi, 0,1,2,3,4,5,6,7,8,9,10,11,12,13,14,15);
}

__device__ __forceinline__ void cvt_store4(__bf16* d, float4 f) {
  d[0] = (__bf16)f.x; d[1] = (__bf16)f.y; d[2] = (__bf16)f.z; d[3] = (__bf16)f.w;
}

// ---------------------------------------------------------------------------
// Router: logits -> integer softmax (Q7 in / Q16 exp) -> top-2 -> combine
// weights + compacted per-expert token lists.
// ---------------------------------------------------------------------------
__global__ __launch_bounds__(256)
void moe_router(const float* __restrict__ x, const float* __restrict__ wg,
                float* __restrict__ combine, int* __restrict__ lists,
                int* __restrict__ counts, int H, int T) {
  int t    = blockIdx.x;
  int wave = threadIdx.x >> 5;
  int lane = threadIdx.x & 31;
  __shared__ float logitS[NEXP];

  const float* xr = x  + (size_t)t    * H;
  const float* wr = wg + (size_t)wave * H;
  float acc = 0.f;
  for (int h = lane; h < H; h += 32) acc += xr[h] * wr[h];
  for (int off = 16; off > 0; off >>= 1) acc += __shfl_xor(acc, off, 32);
  if (lane == 0) logitS[wave] = acc;
  __syncthreads();

  if (threadIdx.x == 0) {
    int li[NEXP], ev[NEXP];
    int mx = (int)0x80000000;
    for (int i = 0; i < NEXP; ++i) {
      li[i] = __float2int_rn(logitS[i] * 128.f);
      mx = (li[i] > mx) ? li[i] : mx;
    }
    int sum = 0;
    for (int i = 0; i < NEXP; ++i) {
      int sh = li[i] - mx;
      if (sh < -1024) sh = -1024;
      ev[i] = (int)(expf((float)sh * (1.f / 128.f)) * 65536.f);  // Q16
      sum += ev[i];
    }
    if (sum < 1) sum = 1;
    int i0 = 0;
    for (int i = 1; i < NEXP; ++i) if (ev[i] > ev[i0]) i0 = i;
    int i1 = (i0 == 0) ? 1 : 0;
    for (int i = 0; i < NEXP; ++i)
      if (i != i0 && ev[i] > ev[i1]) i1 = i;

    float w0 = (float)ev[i0] / (float)sum;
    float w1 = (float)ev[i1] / (float)sum;
    float inv = 1.f / (w0 + w1);
    float* crow = combine + (size_t)t * NEXP;
    for (int i = 0; i < NEXP; ++i) crow[i] = 0.f;
    crow[i0] = w0 * inv;
    crow[i1] = w1 * inv;

    int s0 = atomicAdd(&counts[i0], 1);
    lists[(size_t)i0 * T + s0] = t;
    int s1 = atomicAdd(&counts[i1], 1);
    lists[(size_t)i1 * T + s1] = t;
  }
}

// ---------------------------------------------------------------------------
// GEMM1: h[slot, i] = silu(x @ w1^T) * (x @ w3^T), bf16 WMMA, gathered rows.
// 256 threads (8 waves), block tile 128x64, wave tile 32x32 per output matrix.
// Double-buffered LDS; stage = issue loads -> WMMA -> convert/store.
// Invalid slots are clamped to slot m0 (branch-free gather; their outputs are
// never stored and WMMA rows are independent).
// ---------------------------------------------------------------------------
__global__ __launch_bounds__(256)
void moe_gemm1(const float* __restrict__ x,
               const float* __restrict__ w1e, const float* __restrict__ w3e,
               __bf16* __restrict__ hbuf,
               const int* __restrict__ list,  // offset: e*T + chunkBase
               const int* __restrict__ counts, int e, int chunkBase,
               int H, int I) {
  int cnt = counts[e] - chunkBase;
  if (cnt > TC) cnt = TC;
  int m0 = blockIdx.y * BM;
  if (cnt <= 0 || m0 >= cnt) return;
  int n0  = blockIdx.x * BN;
  int tid = threadIdx.x, lane = tid & 31, wid = tid >> 5;
  int wm  = wid >> 1, wn = wid & 1;          // 4 x 2 waves over 128x64

  __shared__ __attribute__((aligned(16))) __bf16 As [2][BM * LDP];
  __shared__ __attribute__((aligned(16))) __bf16 B1s[2][BN * LDP];
  __shared__ __attribute__((aligned(16))) __bf16 B3s[2][BN * LDP];
  __shared__ int tokS[BM];

  if (tid < BM) {
    int s = m0 + tid;
    // slot m0 is always valid for a live block: clamp invalid rows to it
    tokS[tid] = (s < cnt) ? list[s] : list[m0];
  }
  __syncthreads();

  // per-thread load slots:  A: 4 x float4 (rows arow+32j), B: 2 x float4 each
  int arow = tid >> 3, ac4 = tid & 7;        // rows 0..31 (+32j), cols 0..7
  int aTok[4];
  #pragma unroll
  for (int j = 0; j < 4; ++j) aTok[j] = tokS[arow + j * 32];

  v8f zacc;
  #pragma unroll
  for (int i = 0; i < 8; ++i) zacc[i] = 0.f;
  v8f accG[2][2], accU[2][2];
  #pragma unroll
  for (int a = 0; a < 2; ++a)
    #pragma unroll
    for (int b = 0; b < 2; ++b) { accG[a][b] = zacc; accU[a][b] = zacc; }

  float4 aReg[4], b1Reg[2], b3Reg[2];

  auto issueLoads = [&](int k0) {
    #pragma unroll
    for (int j = 0; j < 4; ++j)
      aReg[j] = *(const float4*)(x + (size_t)aTok[j] * H + k0 + ac4 * 4);
    #pragma unroll
    for (int j = 0; j < 2; ++j) {
      int idx = tid + j * 256;                 // 0..511
      int r = idx >> 3, c = idx & 7;           // r: 0..63
      b1Reg[j] = *(const float4*)(w1e + (size_t)(n0 + r) * H + k0 + c * 4);
      b3Reg[j] = *(const float4*)(w3e + (size_t)(n0 + r) * H + k0 + c * 4);
    }
  };
  auto storeTiles = [&](int buf) {
    #pragma unroll
    for (int j = 0; j < 4; ++j)
      cvt_store4(&As[buf][(arow + j * 32) * LDP + ac4 * 4], aReg[j]);
    #pragma unroll
    for (int j = 0; j < 2; ++j) {
      int idx = tid + j * 256;
      int r = idx >> 3, c = idx & 7;
      cvt_store4(&B1s[buf][r * LDP + c * 4], b1Reg[j]);
      cvt_store4(&B3s[buf][r * LDP + c * 4], b3Reg[j]);
    }
  };
  auto compute = [&](int buf) {
    v16bf af[2];
    #pragma unroll
    for (int mi = 0; mi < 2; ++mi)
      af[mi] = load_a_frag(&As[buf][(wm * 32 + mi * 16) * LDP], lane);
    #pragma unroll
    for (int ni = 0; ni < 2; ++ni) {
      v16bf b1 = load_b_frag(&B1s[buf][(wn * 32 + ni * 16) * LDP], lane);
      v16bf b3 = load_b_frag(&B3s[buf][(wn * 32 + ni * 16) * LDP], lane);
      #pragma unroll
      for (int mi = 0; mi < 2; ++mi) {
        accG[mi][ni] = __builtin_amdgcn_wmma_f32_16x16x32_bf16(
            false, af[mi], false, b1, (short)0, accG[mi][ni], false, false);
        accU[mi][ni] = __builtin_amdgcn_wmma_f32_16x16x32_bf16(
            false, af[mi], false, b3, (short)0, accU[mi][ni], false, false);
      }
    }
  };

  int nk = H / BK;
  issueLoads(0);
  storeTiles(0);
  for (int ks = 0; ks < nk; ++ks) {
    int cur = ks & 1;
    __syncthreads();
    bool pf = (ks + 1 < nk);
    if (pf) issueLoads((ks + 1) * BK);     // loads in flight during WMMA
    compute(cur);
    if (pf) storeTiles(cur ^ 1);
  }

  // epilogue: SwiGLU, store bf16
  #pragma unroll
  for (int mi = 0; mi < 2; ++mi)
    #pragma unroll
    for (int ni = 0; ni < 2; ++ni)
      #pragma unroll
      for (int r = 0; r < 8; ++r) {
        int row = wm * 32 + mi * 16 + ((lane < 16) ? r : r + 8);
        if (m0 + row < cnt) {
          float g = accG[mi][ni][r], u = accU[mi][ni][r];
          float s = g / (1.f + __expf(-g));
          int col = n0 + wn * 32 + ni * 16 + (lane & 15);
          hbuf[(size_t)(m0 + row) * I + col] = (__bf16)(s * u);
        }
      }
}

// ---------------------------------------------------------------------------
// GEMM2: out[tok, :] += combine[tok,e] * (h @ w2^T), f32 accumulate.
// A tile is already bf16 -> async copy global->LDS when available.
// ---------------------------------------------------------------------------
__global__ __launch_bounds__(256)
void moe_gemm2(const __bf16* __restrict__ hbuf,
               const float* __restrict__ w2e,
               const float* __restrict__ combine,
               float* __restrict__ out,
               const int* __restrict__ list,
               const int* __restrict__ counts, int e, int chunkBase,
               int I, int Hdim) {
  int cnt = counts[e] - chunkBase;
  if (cnt > TC) cnt = TC;
  int m0 = blockIdx.y * BM;
  if (cnt <= 0 || m0 >= cnt) return;
  int n0  = blockIdx.x * BN;
  int tid = threadIdx.x, lane = tid & 31, wid = tid >> 5;
  int wm  = wid >> 1, wn = wid & 1;

  __shared__ __attribute__((aligned(16))) __bf16 As[2][BM * LDP];
  __shared__ __attribute__((aligned(16))) __bf16 Bs[2][BN * LDP];
  __shared__ int   tokS[BM];
  __shared__ float cwS [BM];

  if (tid < BM) {
    int s = m0 + tid;
    int tok = (s < cnt) ? list[s] : -1;
    tokS[tid] = tok;
    cwS [tid] = (tok >= 0) ? combine[(size_t)tok * NEXP + e] : 0.f;
  }
  __syncthreads();

  v8f zacc;
  #pragma unroll
  for (int i = 0; i < 8; ++i) zacc[i] = 0.f;
  v8f acc[2][2];
  #pragma unroll
  for (int a = 0; a < 2; ++a)
    #pragma unroll
    for (int b = 0; b < 2; ++b) acc[a][b] = zacc;

  float4 bReg[2];
#ifndef USE_ASYNC_LDS
  v8bf aReg[2];
#endif

  // Rows beyond cnt load stale data: harmless, WMMA rows are independent and
  // those outputs are never stored. hbuf rows < TC are always in-bounds.
  auto issueA = [&](int k0, int buf) {
    #pragma unroll
    for (int j = 0; j < 2; ++j) {
      int idx = tid + j * 256;                 // 0..511
      int r = idx >> 2, c = idx & 3;           // r: 0..127, 16B chunks
      const void* g = hbuf + (size_t)(m0 + r) * I + k0 + c * 8;
#ifdef USE_ASYNC_LDS
      async_ld16(&As[buf][r * LDP + c * 8], g);
#else
      aReg[j] = *(const v8bf*)g;
#endif
    }
  };
  auto issueB = [&](int k0) {
    #pragma unroll
    for (int j = 0; j < 2; ++j) {
      int idx = tid + j * 256;
      int r = idx >> 3, c = idx & 7;           // r: 0..63
      bReg[j] = *(const float4*)(w2e + (size_t)(n0 + r) * I + k0 + c * 4);
    }
  };
  auto storeTiles = [&](int buf) {
#ifndef USE_ASYNC_LDS
    #pragma unroll
    for (int j = 0; j < 2; ++j) {
      int idx = tid + j * 256;
      int r = idx >> 2, c = idx & 3;
      *(v8bf*)&As[buf][r * LDP + c * 8] = aReg[j];
    }
#endif
    #pragma unroll
    for (int j = 0; j < 2; ++j) {
      int idx = tid + j * 256;
      int r = idx >> 3, c = idx & 7;
      cvt_store4(&Bs[buf][r * LDP + c * 4], bReg[j]);
    }
  };
  auto compute = [&](int buf) {
    v16bf af[2];
    #pragma unroll
    for (int mi = 0; mi < 2; ++mi)
      af[mi] = load_a_frag(&As[buf][(wm * 32 + mi * 16) * LDP], lane);
    #pragma unroll
    for (int ni = 0; ni < 2; ++ni) {
      v16bf b = load_b_frag(&Bs[buf][(wn * 32 + ni * 16) * LDP], lane);
      #pragma unroll
      for (int mi = 0; mi < 2; ++mi)
        acc[mi][ni] = __builtin_amdgcn_wmma_f32_16x16x32_bf16(
            false, af[mi], false, b, (short)0, acc[mi][ni], false, false);
    }
  };

  int nk = I / BK;
  issueA(0, 0);
  issueB(0);
  storeTiles(0);
  for (int ks = 0; ks < nk; ++ks) {
    int cur = ks & 1;
#ifdef USE_ASYNC_LDS
    async_fence();                         // our async writes into LDS landed
#endif
    __syncthreads();
    bool pf = (ks + 1 < nk);
    if (pf) { issueA((ks + 1) * BK, cur ^ 1); issueB((ks + 1) * BK); }
    compute(cur);
    if (pf) storeTiles(cur ^ 1);
  }

  #pragma unroll
  for (int mi = 0; mi < 2; ++mi)
    #pragma unroll
    for (int ni = 0; ni < 2; ++ni)
      #pragma unroll
      for (int r = 0; r < 8; ++r) {
        int row = wm * 32 + mi * 16 + ((lane < 16) ? r : r + 8);
        if (m0 + row < cnt) {
          int tok = tokS[row];
          float cw = cwS[row];
          int col = n0 + wn * 32 + ni * 16 + (lane & 15);
          out[(size_t)tok * Hdim + col] += cw * acc[mi][ni][r];
        }
      }
}

// ---------------------------------------------------------------------------
extern "C" void kernel_launch(void* const* d_in, const int* in_sizes, int n_in,
                              void* d_out, int out_size, void* d_ws, size_t ws_size,
                              hipStream_t stream) {
  const float* x  = (const float*)d_in[0];
  const float* wg = (const float*)d_in[1];
  const float* w1 = (const float*)d_in[2];
  const float* w2 = (const float*)d_in[3];
  const float* w3 = (const float*)d_in[4];
  float* out = (float*)d_out;

  const int E = NEXP;
  int H = in_sizes[1] / E;
  long long ih = (long long)in_sizes[2] / E;
  int I = (int)(ih / H);
  int T = in_sizes[0] / H;

  // workspace layout (256B aligned slices)
  char* ws = (char*)d_ws;
  size_t off = 0;
  auto wsAlloc = [&](size_t bytes) -> void* {
    size_t p = (off + 255) & ~(size_t)255;
    off = p + bytes;
    return (void*)(ws + p);
  };
  int*    counts  = (int*)   wsAlloc(E * sizeof(int));
  int*    lists   = (int*)   wsAlloc((size_t)E * T * sizeof(int));
  float*  combine = (float*) wsAlloc((size_t)T * E * sizeof(float));
  __bf16* hbuf    = (__bf16*)wsAlloc((size_t)TC * I * sizeof(__bf16));
  (void)ws_size; (void)n_in; (void)out_size;

  (void)hipMemsetAsync(counts, 0, E * sizeof(int), stream);
  (void)hipMemsetAsync(out, 0, (size_t)T * H * sizeof(float), stream);

  moe_router<<<T, 256, 0, stream>>>(x, wg, combine, lists, counts, H, T);

  int chunks = (T + TC - 1) / TC;
  for (int e = 0; e < E; ++e) {
    for (int c = 0; c < chunks; ++c) {
      const int* lst = lists + (size_t)e * T + (size_t)c * TC;
      dim3 g1(I / BN, TC / BM);
      moe_gemm1<<<g1, 256, 0, stream>>>(x, w1 + (size_t)e * I * H,
                                        w3 + (size_t)e * I * H,
                                        hbuf, lst, counts, e, c * TC, H, I);
      dim3 g2(H / BN, TC / BM);
      moe_gemm2<<<g2, 256, 0, stream>>>(hbuf, w2 + (size_t)e * H * I,
                                        combine, out, lst, counts, e, c * TC, I, H);
    }
  }
}
